// GraphEmbedding_75600014344283
// MI455X (gfx1250) — matrix operand order repaired and checked by reading, not compiled, not run on hardware
//
#include <hip/hip_runtime.h>
#include <hip/hip_bf16.h>

// ---------------------------------------------------------------------------
// TGN temporal graph attention for MI455X (gfx1250, wave32, WMMA).
// GEMMs use v_wmma_f32_16x16x32_bf16. k_in is processed in groups of 80 rows
// (= 4 complete sources x 20 neighbors) so each B-fragment load is reused by
// 5 WMMA M-tiles, cutting L2 weight traffic 5x, and k_in is built only once.
// ---------------------------------------------------------------------------

typedef __attribute__((ext_vector_type(16))) __bf16 v16bf;
typedef __attribute__((ext_vector_type(8)))  float  v8f;

#define K_NBR  20
#define D_FEAT 128
#define T_DIM  128
#define DQ     256    // D + T
#define DK     384    // D + T + D
#define ROWS   16     // source rows per workgroup
#define NTH    256
#define NWAVES 8
#define GROUP_ROWS 80 // 4 sources x 20 neighbors
#define NGROUPS 4
#define MTILES  5     // 80 / 16

// dynamic-LDS layout (bytes)
#define SM_KIN  0                        // [80][384] bf16 (also q_in / fc1-in alias)
#define SM_P    (SM_KIN + 80*384*2)      // [80][256] bf16  K-proj / V-proj
#define SM_QB   (SM_P   + 80*256*2)      // [16][256] bf16  Q, later attn-out
#define SM_O    (SM_QB  + 16*256*2)      // [16][256] f32   Wo / fc1 / fc2 out
#define SM_SRC  (SM_O   + 16*256*4)      // [16][128] bf16
#define SM_SC   (SM_SRC + 16*128*2)      // [16][20][2] f32 scores -> attn
#define SM_H1   (SM_SC  + 16*20*2*4)     // [16][128] bf16
#define SM_MASK (SM_H1  + 16*128*2)      // [16][20] u8
#define SM_ALLM (SM_MASK + 320)          // [16] int
#define SM_TS   (SM_ALLM + 64)           // [16] f32
#define SM_DT   (SM_TS  + 64)            // [80] f32
#define SM_TOTAL (SM_DT + 320)           // ~138.5 KB

__device__ __forceinline__ __bf16 f2bf(float f) {
    unsigned u = __builtin_bit_cast(unsigned, f);
    u += 0x7FFFu + ((u >> 16) & 1u);  // round-to-nearest-even
    unsigned short h = (unsigned short)(u >> 16);
    return __builtin_bit_cast(__bf16, h);
}
__device__ __forceinline__ float bf2f(__bf16 b) {
    unsigned u = ((unsigned)__builtin_bit_cast(unsigned short, b)) << 16;
    return __builtin_bit_cast(float, u);
}

// A fragment (16x32 bf16) from row-major LDS tile. contiguous k order used here (not the ISA table order):
// lane m=L%16 is row M; element e -> K = 16g + e (matches load_b).
__device__ __forceinline__ v16bf load_a(const __bf16* A, int lda, int kb, int lane) {
    int m = lane & 15;
    int g = (lane >> 4) & 1;
    const __bf16* r = A + m * lda + kb;
    v16bf a;
#pragma unroll
    for (int j = 0; j < 4; ++j) {
        a[2 * j + 0] = r[16 * g + 2 * j + 0];
        a[2 * j + 1] = r[16 * g + 2 * j + 1];
        a[8 + 2 * j + 0] = r[16 * g + 8 + 2 * j + 0];
        a[8 + 2 * j + 1] = r[16 * g + 8 + 2 * j + 1];
    }
    return a;
}

// B fragment (32x16 bf16) from pre-transposed weights WT[Nd][Kd].
// lane n=L%16 is column N; element e -> K = 16g + e  => contiguous 32B.
__device__ __forceinline__ v16bf load_b(const __bf16* WT, int ldk, int ncb, int kb, int lane) {
    int n = lane & 15;
    int g = (lane >> 4) & 1;
    const __bf16* p = WT + (size_t)(ncb + n) * ldk + kb + 16 * g;
    return *(const v16bf*)p;
}

// Single 16-row GEMM: D[16 x 16*Ntiles] = A[16 x Kdim] @ WT^T + bias.
template <bool BF16OUT>
__device__ __forceinline__ void gemm_tile(const __bf16* A, int Kdim,
                                          const __bf16* WT, const float* bias,
                                          void* Out, int ldo, int Ntiles,
                                          int wave, int lane) {
    for (int nt = wave; nt < Ntiles; nt += NWAVES) {
        v8f acc = {};
        for (int kb = 0; kb < Kdim; kb += 32) {
            v16bf a = load_a(A, Kdim, kb, lane);
            v16bf b = load_b(WT, Kdim, nt * 16, kb, lane);
            acc = __builtin_amdgcn_wmma_f32_16x16x32_bf16(
                false, a, false, b, (short)0, acc, false, false);
        }
        int n = lane & 15, g = (lane >> 4) & 1;
        float bv = bias[nt * 16 + n];
#pragma unroll
        for (int r = 0; r < 8; ++r) {
            float v = acc[r] + bv;
            size_t off = (size_t)(r + 8 * g) * ldo + nt * 16 + n;  // C/D row=r+8g
            if (BF16OUT) ((__bf16*)Out)[off] = f2bf(v);
            else         ((float*)Out)[off]  = v;
        }
    }
}

// 80-row GEMM: one B-fragment load feeds 5 M-tile WMMAs.
__device__ __forceinline__ void gemm_group(const __bf16* A, const __bf16* WT,
                                           const float* bias, __bf16* Out,
                                           int wave, int lane) {
    for (int nt = wave; nt < DQ / 16; nt += NWAVES) {
        v8f zero = {};
        v8f acc[MTILES];
#pragma unroll
        for (int m = 0; m < MTILES; ++m) acc[m] = zero;
        for (int kb = 0; kb < DK; kb += 32) {
            v16bf b = load_b(WT, DK, nt * 16, kb, lane);
#pragma unroll
            for (int m = 0; m < MTILES; ++m) {
                v16bf a = load_a(A + (size_t)m * 16 * DK, DK, kb, lane);
                acc[m] = __builtin_amdgcn_wmma_f32_16x16x32_bf16(
                    false, a, false, b, (short)0, acc[m], false, false);
            }
        }
        int n = lane & 15, g = (lane >> 4) & 1;
        float bv = bias[nt * 16 + n];
#pragma unroll
        for (int m = 0; m < MTILES; ++m)
#pragma unroll
            for (int r = 0; r < 8; ++r)
                Out[(size_t)(m * 16 + r + 8 * g) * DQ + nt * 16 + n] =
                    f2bf(acc[m][r] + bv);
    }
}

// Transpose fp32 W[2][Kd][Nd] -> bf16 WT[2][Nd][Kd].
__global__ void tgn_wprep(const float* __restrict__ src, __bf16* __restrict__ dst,
                          int Kd, int Nd) {
    long total = 2L * Kd * Nd;
    for (long i = (long)blockIdx.x * blockDim.x + threadIdx.x; i < total;
         i += (long)gridDim.x * blockDim.x) {
        int l = (int)(i / ((long)Kd * Nd));
        int rem = (int)(i % ((long)Kd * Nd));
        int k = rem / Nd, n = rem % Nd;
        dst[(long)l * Kd * Nd + (long)n * Kd + k] = f2bf(src[i]);
    }
}

__global__ __launch_bounds__(NTH) void tgn_aggregate(
    const float* __restrict__ node_features, const float* __restrict__ edge_feat,
    const float* __restrict__ memory_tab, const float* __restrict__ timestamps,
    const float* __restrict__ etimes, const float* __restrict__ tw,
    const float* __restrict__ tb,
    const int* __restrict__ nodes, const int* __restrict__ nbrs,
    const int* __restrict__ eidx,
    const float* __restrict__ src_in, const float* __restrict__ neigh_in,
    const __bf16* __restrict__ WqT, const __bf16* __restrict__ WkT,
    const __bf16* __restrict__ WvT, const __bf16* __restrict__ WoT,
    const __bf16* __restrict__ fc1T, const __bf16* __restrict__ fc2T,
    const float* __restrict__ bq, const float* __restrict__ bk,
    const float* __restrict__ bv_, const float* __restrict__ bo,
    const float* __restrict__ fc1b, const float* __restrict__ fc2b,
    float* __restrict__ out, int tsDiv, int srcMode, int neighMode) {
    extern __shared__ char smem[];
    __bf16 (*sh_kin)[DK]      = (__bf16(*)[DK])(smem + SM_KIN);
    __bf16 (*sh_qin)[DQ]      = (__bf16(*)[DQ])(smem + SM_KIN);   // alias (pre-group)
    __bf16 (*sh_P)[DQ]        = (__bf16(*)[DQ])(smem + SM_P);
    __bf16 (*sh_Qb)[DQ]       = (__bf16(*)[DQ])(smem + SM_QB);
    float  (*sh_O)[DQ]        = (float (*)[DQ])(smem + SM_O);
    __bf16 (*sh_src)[D_FEAT]  = (__bf16(*)[D_FEAT])(smem + SM_SRC);
    float  (*sh_sc)[K_NBR][2] = (float (*)[K_NBR][2])(smem + SM_SC);
    __bf16 (*sh_h1)[D_FEAT]   = (__bf16(*)[D_FEAT])(smem + SM_H1);
    unsigned char (*sh_mask)[K_NBR] = (unsigned char (*)[K_NBR])(smem + SM_MASK);
    int*   sh_allm = (int*)(smem + SM_ALLM);
    float* sh_ts   = (float*)(smem + SM_TS);
    float* sh_dt   = (float*)(smem + SM_DT);

    const int r0 = blockIdx.x * ROWS;
    const int tid = threadIdx.x;
    const int wave = tid >> 5, lane = tid & 31;

    // ---- init: timestamps, padding mask ----
    if (tid < ROWS) sh_ts[tid] = timestamps[(r0 + tid) / tsDiv];
    for (int idx = tid; idx < ROWS * K_NBR; idx += NTH) {
        int s = idx / K_NBR, j = idx % K_NBR;
        sh_mask[s][j] = (nbrs[(size_t)(r0 + s) * K_NBR + j] == 0) ? 1 : 0;
    }
    __syncthreads();
    if (tid < ROWS) {
        int all = 1;
        for (int j = 0; j < K_NBR; ++j) all &= (int)sh_mask[tid][j];
        sh_allm[tid] = all;
    }
    // ---- q_in = [src_feat | cos(tb)]  (float4 build) ----
    for (int e = tid; e < ROWS * (DQ / 4); e += NTH) {
        int s = e / (DQ / 4), c4 = (e % (DQ / 4)) * 4;
        float4 v;
        if (c4 < D_FEAT) {
            if (srcMode == 0) {
                int nd = nodes[r0 + s];
                float4 a = *(const float4*)(memory_tab + (size_t)nd * D_FEAT + c4);
                float4 b = *(const float4*)(node_features + (size_t)nd * D_FEAT + c4);
                v = make_float4(a.x + b.x, a.y + b.y, a.z + b.z, a.w + b.w);
            } else {
                v = *(const float4*)(src_in + (size_t)(r0 + s) * D_FEAT + c4);
            }
            sh_src[s][c4 + 0] = f2bf(v.x); sh_src[s][c4 + 1] = f2bf(v.y);
            sh_src[s][c4 + 2] = f2bf(v.z); sh_src[s][c4 + 3] = f2bf(v.w);
        } else {
            int t = c4 - D_FEAT;
            float4 b4 = *(const float4*)(tb + t);
            v = make_float4(cosf(b4.x), cosf(b4.y), cosf(b4.z), cosf(b4.w));
        }
        sh_qin[s][c4 + 0] = f2bf(v.x); sh_qin[s][c4 + 1] = f2bf(v.y);
        sh_qin[s][c4 + 2] = f2bf(v.z); sh_qin[s][c4 + 3] = f2bf(v.w);
    }
    __syncthreads();

    // ---- Q = q_in @ Wq + bq  (bf16 out) ----
    gemm_tile<true>(&sh_qin[0][0], DQ, WqT, bq, &sh_Qb[0][0], DQ, DQ / 16, wave, lane);
    __syncthreads();

    // ---- attention output accumulators (registers) ----
    float oacc[16];
#pragma unroll
    for (int i = 0; i < 16; ++i) oacc[i] = 0.f;
    const int my_s = tid >> 4;
    const int c0 = (tid & 15) * 16;
    const int my_h = c0 >> 7;

    // ---- per-group: build k_in, K-proj, scores, softmax, V-proj, accumulate --
    for (int g = 0; g < NGROUPS; ++g) {
        if (tid < GROUP_ROWS) {
            int grow = g * GROUP_ROWS + tid;
            int s = grow / K_NBR, j = grow % K_NBR;
            sh_dt[tid] = sh_ts[s] - etimes[(size_t)(r0 + s) * K_NBR + j];
        }
        __syncthreads();
        for (int e = tid; e < GROUP_ROWS * (DK / 4); e += NTH) {
            int i = e / (DK / 4), c4 = (e % (DK / 4)) * 4;
            int grow = g * GROUP_ROWS + i;
            int s = grow / K_NBR, j = grow % K_NBR;
            int row = r0 + s;
            float4 v;
            if (c4 < D_FEAT) {
                if (neighMode == 0) {
                    int nd = nbrs[(size_t)row * K_NBR + j];
                    float4 a = *(const float4*)(memory_tab + (size_t)nd * D_FEAT + c4);
                    float4 b = *(const float4*)(node_features + (size_t)nd * D_FEAT + c4);
                    v = make_float4(a.x + b.x, a.y + b.y, a.z + b.z, a.w + b.w);
                } else {
                    v = *(const float4*)(neigh_in + ((size_t)row * K_NBR + j) * D_FEAT + c4);
                }
            } else if (c4 < D_FEAT + T_DIM) {
                int t = c4 - D_FEAT;
                float dt = sh_dt[i];
                float4 w4 = *(const float4*)(tw + t);
                float4 b4 = *(const float4*)(tb + t);
                v = make_float4(cosf(dt * w4.x + b4.x), cosf(dt * w4.y + b4.y),
                                cosf(dt * w4.z + b4.z), cosf(dt * w4.w + b4.w));
            } else {
                int d = c4 - (D_FEAT + T_DIM);
                int ee = eidx[(size_t)row * K_NBR + j];
                v = *(const float4*)(edge_feat + (size_t)ee * D_FEAT + d);
            }
            sh_kin[i][c4 + 0] = f2bf(v.x); sh_kin[i][c4 + 1] = f2bf(v.y);
            sh_kin[i][c4 + 2] = f2bf(v.z); sh_kin[i][c4 + 3] = f2bf(v.w);
        }
        __syncthreads();

        // K-projection for 80 rows (5x B reuse)
        gemm_group(&sh_kin[0][0], WkT, bk, &sh_P[0][0], wave, lane);
        __syncthreads();

        // raw scores: 80 rows x 2 heads, dot over dh=128
        for (int task = tid; task < 2 * GROUP_ROWS; task += NTH) {
            int i = task >> 1, h = task & 1;
            int grow = g * GROUP_ROWS + i;
            int s = grow / K_NBR, j = grow % K_NBR;
            float acc = 0.f;
            for (int d = 0; d < 128; ++d)
                acc += bf2f(sh_Qb[s][h * 128 + d]) * bf2f(sh_P[i][h * 128 + d]);
            sh_sc[s][j][h] = acc;
        }
        __syncthreads();

        // softmax for this group's 4 sources (TGN all-masked trick)
        if (tid < 8) {
            int s = g * 4 + (tid >> 1), h = tid & 1;
            const float scale = 0.0883883476483184f;  // 1/sqrt(128)
            int allm = sh_allm[s];
            float sc[K_NBR];
            float mx = -3.4e38f;
            for (int j = 0; j < K_NBR; ++j) {
                int eff = sh_mask[s][j] && !(allm && j == 0);
                float x = eff ? -1e9f : sh_sc[s][j][h] * scale;
                sc[j] = x;
                mx = fmaxf(mx, x);
            }
            float sum = 0.f;
            for (int j = 0; j < K_NBR; ++j) { sc[j] = expf(sc[j] - mx); sum += sc[j]; }
            float inv = 1.f / sum;
            for (int j = 0; j < K_NBR; ++j) sh_sc[s][j][h] = sc[j] * inv;
        }
        __syncthreads();

        // V-projection for the same 80 rows
        gemm_group(&sh_kin[0][0], WvT, bv_, &sh_P[0][0], wave, lane);
        __syncthreads();

        // attn-weighted sum into registers (threads whose source is in group)
        if ((my_s >> 2) == g) {
            int ibase = (my_s - g * 4) * K_NBR;
            for (int j = 0; j < K_NBR; ++j) {
                float w = sh_sc[my_s][j][my_h];
                int rrow = ibase + j;
#pragma unroll
                for (int cc = 0; cc < 16; ++cc)
                    oacc[cc] += w * bf2f(sh_P[rrow][c0 + cc]);
            }
        }
        __syncthreads();
    }

    // ---- o -> bf16 (reuse Q buffer) ----
#pragma unroll
    for (int cc = 0; cc < 16; ++cc) sh_Qb[my_s][c0 + cc] = f2bf(oacc[cc]);
    __syncthreads();

    // ---- o @ Wo + bo ----
    gemm_tile<false>(&sh_Qb[0][0], DQ, WoT, bo, &sh_O[0][0], DQ, DQ / 16, wave, lane);
    __syncthreads();

    // ---- fc1 input: [masked o (256) | src_feat (128)] ----
    for (int idx = tid; idx < ROWS * DK; idx += NTH) {
        int s = idx / DK, c = idx % DK;
        __bf16 v;
        if (c < DQ) v = f2bf(sh_allm[s] ? 0.f : sh_O[s][c]);
        else        v = sh_src[s][c - DQ];
        sh_kin[s][c] = v;
    }
    __syncthreads();
    gemm_tile<false>(&sh_kin[0][0], DK, fc1T, fc1b, &sh_O[0][0], DQ, D_FEAT / 16, wave, lane);
    __syncthreads();
    for (int idx = tid; idx < ROWS * D_FEAT; idx += NTH) {
        int s = idx / D_FEAT, d = idx % D_FEAT;
        sh_h1[s][d] = f2bf(fmaxf(sh_O[s][d], 0.f));  // ReLU
    }
    __syncthreads();
    gemm_tile<false>(&sh_h1[0][0], D_FEAT, fc2T, fc2b, &sh_O[0][0], DQ, D_FEAT / 16, wave, lane);
    __syncthreads();
    for (int e = tid; e < ROWS * (D_FEAT / 4); e += NTH) {
        int s = e / (D_FEAT / 4), d = (e % (D_FEAT / 4)) * 4;
        float4 v = make_float4(sh_O[s][d], sh_O[s][d + 1], sh_O[s][d + 2], sh_O[s][d + 3]);
        *(float4*)(out + (size_t)(r0 + s) * D_FEAT + d) = v;
    }
}

extern "C" void kernel_launch(void* const* d_in, const int* in_sizes, int n_in,
                              void* d_out, int out_size, void* d_ws, size_t ws_size,
                              hipStream_t stream) {
    (void)n_in; (void)out_size; (void)ws_size;
    const float* node_features = (const float*)d_in[0];
    const float* edge_feat     = (const float*)d_in[1];
    const float* memory_tab    = (const float*)d_in[2];
    const float* timestamps    = (const float*)d_in[3];
    const float* etimes_src    = (const float*)d_in[4];
    const float* etimes_nbr    = (const float*)d_in[5];
    const float* tw            = (const float*)d_in[6];
    const float* tb            = (const float*)d_in[7];
    const float* Wq  = (const float*)d_in[8];
    const float* bq  = (const float*)d_in[9];
    const float* Wk  = (const float*)d_in[10];
    const float* bk  = (const float*)d_in[11];
    const float* Wv  = (const float*)d_in[12];
    const float* bv  = (const float*)d_in[13];
    const float* Wo  = (const float*)d_in[14];
    const float* bo  = (const float*)d_in[15];
    const float* fc1w = (const float*)d_in[16];
    const float* fc1b = (const float*)d_in[17];
    const float* fc2w = (const float*)d_in[18];
    const float* fc2b = (const float*)d_in[19];
    const int* source_nodes = (const int*)d_in[20];
    const int* nbrs_src = (const int*)d_in[21];
    const int* eidx_src = (const int*)d_in[22];
    const int* nbrs_nbr = (const int*)d_in[23];
    const int* eidx_nbr = (const int*)d_in[24];

    const int B = in_sizes[3];  // 2048
    char* ws = (char*)d_ws;
    const size_t OFF_WQ  = 0;
    const size_t OFF_WK  = OFF_WQ + 2ul * 256 * 256 * 2;
    const size_t OFF_WV  = OFF_WK + 2ul * 256 * 384 * 2;
    const size_t OFF_WO  = OFF_WV + 2ul * 256 * 384 * 2;
    const size_t OFF_F1  = OFF_WO + 2ul * 256 * 256 * 2;
    const size_t OFF_F2  = OFF_F1 + 2ul * 128 * 384 * 2;
    const size_t OFF_SRC = OFF_F2 + 2ul * 128 * 128 * 2;
    const size_t OFF_NEI = OFF_SRC + (size_t)B * 128 * 4;

    __bf16* WqT  = (__bf16*)(ws + OFF_WQ);
    __bf16* WkT  = (__bf16*)(ws + OFF_WK);
    __bf16* WvT  = (__bf16*)(ws + OFF_WV);
    __bf16* WoT  = (__bf16*)(ws + OFF_WO);
    __bf16* fc1T = (__bf16*)(ws + OFF_F1);
    __bf16* fc2T = (__bf16*)(ws + OFF_F2);
    float* src_conv  = (float*)(ws + OFF_SRC);
    float* neigh_emb = (float*)(ws + OFF_NEI);

    tgn_wprep<<<512, 256, 0, stream>>>(Wq,   WqT,  256, 256);
    tgn_wprep<<<768, 256, 0, stream>>>(Wk,   WkT,  384, 256);
    tgn_wprep<<<768, 256, 0, stream>>>(Wv,   WvT,  384, 256);
    tgn_wprep<<<512, 256, 0, stream>>>(Wo,   WoT,  256, 256);
    tgn_wprep<<<384, 256, 0, stream>>>(fc1w, fc1T, 384, 128);
    tgn_wprep<<<128, 256, 0, stream>>>(fc2w, fc2T, 128, 128);

    const size_t LQ = 256 * 256, LKV = 384 * 256, L1 = 384 * 128, L2 = 128 * 128;

    // call 1: layer-1 on sources -> src_conv
    tgn_aggregate<<<B / ROWS, NTH, SM_TOTAL, stream>>>(
        node_features, edge_feat, memory_tab, timestamps, etimes_src, tw, tb,
        source_nodes, nbrs_src, eidx_src, nullptr, nullptr,
        WqT, WkT, WvT, WoT, fc1T, fc2T,
        bq, bk, bv, bo, fc1b, fc2b,
        src_conv, /*tsDiv=*/1, /*srcMode=*/0, /*neighMode=*/0);

    // call 2: layer-1 on sources' neighbors -> neigh_emb (ts index = row/K)
    tgn_aggregate<<<(B * K_NBR) / ROWS, NTH, SM_TOTAL, stream>>>(
        node_features, edge_feat, memory_tab, timestamps, etimes_nbr, tw, tb,
        nbrs_src /*flattened node ids*/, nbrs_nbr, eidx_nbr, nullptr, nullptr,
        WqT, WkT, WvT, WoT, fc1T, fc2T,
        bq, bk, bv, bo, fc1b, fc2b,
        neigh_emb, /*tsDiv=*/K_NBR, /*srcMode=*/0, /*neighMode=*/0);

    // call 3: layer-2 aggregation -> d_out
    tgn_aggregate<<<B / ROWS, NTH, SM_TOTAL, stream>>>(
        node_features, edge_feat, memory_tab, timestamps, etimes_src, tw, tb,
        nullptr, nbrs_src, eidx_src, src_conv, neigh_emb,
        WqT + LQ, WkT + LKV, WvT + LKV, WoT + LQ, fc1T + L1, fc2T + L2,
        bq + 256, bk + 256, bv + 256, bo + 256, fc1b + 128, fc2b + 128,
        (float*)d_out, /*tsDiv=*/1, /*srcMode=*/1, /*neighMode=*/1);
}
